// MotionCompensation_13975823581679
// MI455X (gfx1250) — compile-verified, matrix operand order checked by reading, not run on hardware
//
#include <hip/hip_runtime.h>

// ---------------------------------------------------------------------------
// MotionCompensation for MI455X (gfx1250, wave32).
// Dense 3x3 convs -> implicit GEMM on v_wmma_f32_16x16x32_bf16, fully
// unrolled K-loop, fragments loaded as 2x ds_load_b128 each (contiguous per
// the ISA 16-bit A/B layout), bank-conflict-free padded LDS pitches
// (pitch words % 4 == 0 and pitch/4 odd -> 16 lanes x 4 banks tile all 64).
// Deformable conv -> VALU bilinear gather with grouped weights in LDS.
// ---------------------------------------------------------------------------

typedef __bf16 bf16;
typedef __attribute__((ext_vector_type(16))) __bf16 v16bf;
typedef __attribute__((ext_vector_type(8)))  float  v8f;

#define IMG_H 256
#define IMG_W 256
#define IMG_B 2

// ---------------------------------------------------------------------------
// conv3x3 SAME, NCHW, via WMMA implicit GEMM.
//   GEMM: D[M=out-ch 16][N=pixel 16] += A[M][K] * B[K][N],  K = 9*CIN, tap-major
//   (k = tap*CIN + cin): each K=32 chunk is one tap / 32 consecutive cin.
//   A = weights (bf16 LDS, [m][KPAD]),  B = input halo (bf16 LDS,
//   [row][x][CSTR]).  Per the ISA 16-bit layout, a lane's fragment is two
//   contiguous 16-byte blocks at half-offsets {8h, 16+8h} -> ds_load_b128.
// Block: 256 threads = 8 waves, each wave owns 16 consecutive x pixels.
// CIN1 < CIN => channel-concat of in0 (CIN1 ch) and in1 (CIN-CIN1 ch).
// ---------------------------------------------------------------------------
template <int CIN, int CIN1, int COUT, bool RELU, bool ADD_PRED>
__global__ __launch_bounds__(256) void conv3x3_wmma_kernel(
    const float* __restrict__ in0, const float* __restrict__ in1,
    const float* __restrict__ wgt,   // (COUT, CIN, 3, 3) OIHW
    const float* __restrict__ bias,  // (COUT)
    const float* __restrict__ pred,  // residual add source (ADD_PRED)
    float* __restrict__ out)         // (B, COUT, H, W)
{
    constexpr int H = IMG_H, W = IMG_W;
    constexpr int WT = 128;            // pixels per block (8 waves x 16)
    constexpr int XP = WT + 2;         // 130 valid halo columns
    constexpr int XPITCH = 132;        // LDS x pitch (pixels)
    constexpr int CSTR = CIN + 8;      // halves per pixel: 16B-aligned, /8 odd
    constexpr int K = 9 * CIN;
    constexpr int KPAD = K + 8;        // weight row pitch: 16B-aligned, /8 odd
    constexpr int KCH = K / 32;        // 18 (CIN=64) or 36 (CIN=128)

    extern __shared__ char smem_raw[];
    bf16* sIn = (bf16*)smem_raw;                                   // [3][XPITCH][CSTR]
    bf16* sW  = (bf16*)(smem_raw + (size_t)3 * XPITCH * CSTR * 2); // [16][KPAD]

    const int tid = threadIdx.x;
    const int b   = blockIdx.z;
    const int y   = blockIdx.y;
    const int xb  = blockIdx.x * WT;

    // ---- stage input halo rows y-1..y+1, x in [xb-1, xb+WT] as bf16 ----
    for (int e = tid; e < 3 * XP * CIN; e += 256) {
        const int x   = e % XP;                 // coalesced along x
        const int row = (e / XP) % 3;
        const int c   = e / (XP * 3);
        const int gy  = y + row - 1;
        const int gx  = xb + x - 1;
        float v = 0.0f;
        if (gy >= 0 && gy < H && gx >= 0 && gx < W) {
            if constexpr (CIN1 < CIN) {
                if (c < CIN1)
                    v = in0[(((size_t)b * CIN1 + c) * H + gy) * W + gx];
                else
                    v = in1[(((size_t)b * (CIN - CIN1) + (c - CIN1)) * H + gy) * W + gx];
            } else {
                v = in0[(((size_t)b * CIN + c) * H + gy) * W + gx];
            }
        }
        sIn[(row * XPITCH + x) * CSTR + c] = (bf16)v;
    }
    __syncthreads();

    const int wave = tid >> 5;
    const int lane = tid & 31;
    const int half = lane >> 4;
    const int l16  = lane & 15;
    const int xloc = wave * 16 + l16;          // this lane's pixel (N index)

    for (int nt = 0; nt < COUT / 16; ++nt) {
        // ---- stage 16 output channels of weights: sW[m][k], k = t*CIN + c ----
        for (int e = tid; e < 16 * K; e += 256) {
            const int m = e / K;
            const int k = e % K;
            const int t = k / CIN;
            const int c = k % CIN;
            sW[m * KPAD + k] = (bf16)wgt[((size_t)(nt * 16 + m) * CIN + c) * 9 + t];
        }
        __syncthreads();

        // Warm L2/WGP$ for the *next* n-tile's weights (global_prefetch_b8).
        if (nt + 1 < COUT / 16) {
            const float* nw = wgt + (size_t)(nt + 1) * 16 * CIN * 9;
            for (int p = tid; p * 64 < 16 * CIN * 9; p += 256)
                __builtin_prefetch(nw + (size_t)p * 64, 0, 1);
        }

        // ---- fully unrolled K loop: 2x b128 per fragment, dual accs ----
        v8f acc0 = {}, acc1 = {};
#pragma unroll
        for (int kc = 0; kc < KCH; ++kc) {
            const int t   = (kc * 32) / CIN;    // compile-time after unroll
            const int c0  = (kc * 32) % CIN;
            const int row = t / 3;
            const int xx  = xloc + (t % 3);     // halo x, 0..129
            // lane fragment bases (halves); both 16B-aligned:
            const int aBase = l16 * KPAD + kc * 32 + half * 8;
            const int bBase = (row * XPITCH + xx) * CSTR + c0 + half * 8;

            union { v16bf h; uint4 q[2]; } A, Bm;
            A.q[0]  = *(const uint4*)(sW + aBase);
            A.q[1]  = *(const uint4*)(sW + aBase + 16);
            Bm.q[0] = *(const uint4*)(sIn + bBase);
            Bm.q[1] = *(const uint4*)(sIn + bBase + 16);

            if (kc & 1)
                acc1 = __builtin_amdgcn_wmma_f32_16x16x32_bf16(
                    false, A.h, false, Bm.h, (short)0, acc1, false, false);
            else
                acc0 = __builtin_amdgcn_wmma_f32_16x16x32_bf16(
                    false, A.h, false, Bm.h, (short)0, acc0, false, false);
        }
        __syncthreads();   // protect sW before next n-tile overwrites it

        // ---- epilogue: D VGPR r -> M = r + 8*half (channel), N = l16 (pixel)
        const int gx = xb + xloc;
#pragma unroll
        for (int r = 0; r < 8; ++r) {
            const int ch = nt * 16 + r + 8 * half;
            float v = acc0[r] + acc1[r] + bias[ch];
            if constexpr (RELU) v = fmaxf(v, 0.0f);
            const size_t oi = (((size_t)b * COUT + ch) * H + y) * W + gx;
            if constexpr (ADD_PRED) v += pred[oi];
            out[oi] = v;
        }
    }
}

// ---------------------------------------------------------------------------
// Deformable conv, G=8 groups, Cg=8, 3x3. One thread per (b, g, y, x):
// 9 taps x 8 in-ch bilinear samples, 8 out-ch accumulators, grouped weights
// (64x8x9 f32 = 18.4 KB) staged once in LDS.
// ---------------------------------------------------------------------------
__global__ __launch_bounds__(256) void deform_conv_kernel(
    const float* __restrict__ ref,     // (B, 64, H, W)
    const float* __restrict__ offset,  // (B, 144, H, W)
    const float* __restrict__ wdef,    // (64, 8, 3, 3)
    const float* __restrict__ bdef,    // (64)
    float* __restrict__ pred)          // (B, 64, H, W)
{
    constexpr int H = IMG_H, W = IMG_W;
    __shared__ float sw[64 * 8 * 9];
    for (int e = threadIdx.x; e < 64 * 8 * 9; e += 256) sw[e] = wdef[e];
    __syncthreads();

    const int idx = blockIdx.x * 256 + threadIdx.x;
    if (idx >= IMG_B * 8 * H * W) return;
    const int x = idx % W;
    const int y = (idx / W) % H;
    const int g = (idx / (W * H)) % 8;
    const int b = idx / (W * H * 8);

    const size_t hw      = (size_t)H * W;
    const size_t offBase = (size_t)b * 144 * hw + (size_t)y * W + x;
    const size_t refBase = ((size_t)b * 64 + g * 8) * hw;

    float acc[8];
#pragma unroll
    for (int o = 0; o < 8; ++o) acc[o] = 0.0f;

    for (int t = 0; t < 9; ++t) {
        const int ti = t / 3, tj = t % 3;
        const int chD = (g * 9 + t) * 2;
        const float dy = offset[offBase + (size_t)chD * hw];
        const float dx = offset[offBase + (size_t)(chD + 1) * hw];
        const float py = (float)y - 1.0f + (float)ti + dy;
        const float px = (float)x - 1.0f + (float)tj + dx;
        const float fy0 = floorf(py), fx0 = floorf(px);
        const float wy1 = py - fy0, wx1 = px - fx0;
        const float wy0 = 1.0f - wy1, wx0 = 1.0f - wx1;
        const int iy0 = (int)fy0, ix0 = (int)fx0;
        const bool vy0 = (iy0 >= 0) && (iy0 < H);
        const bool vy1 = (iy0 + 1 >= 0) && (iy0 + 1 < H);
        const bool vx0 = (ix0 >= 0) && (ix0 < W);
        const bool vx1 = (ix0 + 1 >= 0) && (ix0 + 1 < W);
        const int cy0 = min(max(iy0, 0), H - 1);
        const int cy1 = min(max(iy0 + 1, 0), H - 1);
        const int cx0 = min(max(ix0, 0), W - 1);
        const int cx1 = min(max(ix0 + 1, 0), W - 1);
        const float m00 = (vy0 && vx0) ? wy0 * wx0 : 0.0f;
        const float m01 = (vy0 && vx1) ? wy0 * wx1 : 0.0f;
        const float m10 = (vy1 && vx0) ? wy1 * wx0 : 0.0f;
        const float m11 = (vy1 && vx1) ? wy1 * wx1 : 0.0f;
        const int p00 = cy0 * W + cx0, p01 = cy0 * W + cx1;
        const int p10 = cy1 * W + cx0, p11 = cy1 * W + cx1;

        for (int c = 0; c < 8; ++c) {
            const float* img = ref + refBase + (size_t)c * hw;
            const float s = m00 * img[p00] + m01 * img[p01] +
                            m10 * img[p10] + m11 * img[p11];
#pragma unroll
            for (int o = 0; o < 8; ++o)
                acc[o] = fmaf(s, sw[((g * 8 + o) * 8 + c) * 9 + t], acc[o]);
        }
    }
#pragma unroll
    for (int o = 0; o < 8; ++o)
        pred[((size_t)b * 64 + g * 8 + o) * hw + (size_t)y * W + x] =
            acc[o] + bdef[g * 8 + o];
}

// ---------------------------------------------------------------------------
extern "C" void kernel_launch(void* const* d_in, const int* in_sizes, int n_in,
                              void* d_out, int out_size, void* d_ws, size_t ws_size,
                              hipStream_t stream) {
    (void)in_sizes; (void)n_in; (void)out_size; (void)ws_size;
    const float* ref   = (const float*)d_in[0];
    const float* mot   = (const float*)d_in[1];
    const float* w_off = (const float*)d_in[2];
    const float* b_off = (const float*)d_in[3];
    const float* w_def = (const float*)d_in[4];
    const float* b_def = (const float*)d_in[5];
    const float* w_r1  = (const float*)d_in[6];
    const float* b_r1  = (const float*)d_in[7];
    const float* w_r2  = (const float*)d_in[8];
    const float* b_r2  = (const float*)d_in[9];
    float* out = (float*)d_out;

    const size_t HWsz = (size_t)IMG_H * IMG_W;
    float* offs = (float*)d_ws;                      // B*144*H*W
    float* pred = offs + (size_t)IMG_B * 144 * HWsz; // B*64*H*W
    float* hbuf = pred + (size_t)IMG_B * 64 * HWsz;  // B*64*H*W

    const dim3 blk(256);
    const dim3 cgrid(IMG_W / 128, IMG_H, IMG_B);
    auto ldsBytes = [](int cin) {
        // input halo [3][132][cin+8] + weights [16][9*cin+8], bf16
        return (size_t)(3 * 132 * (cin + 8) + 16 * (9 * cin + 8)) * 2;
    };

    // 1) offset = conv3x3(motion_fields) : 64 -> 144
    conv3x3_wmma_kernel<64, 64, 144, false, false>
        <<<cgrid, blk, ldsBytes(64), stream>>>(mot, nullptr, w_off, b_off,
                                               nullptr, offs);
    // 2) pred = deform_conv(ref, offset)
    deform_conv_kernel<<<(IMG_B * 8 * IMG_H * IMG_W) / 256, blk, 0, stream>>>(
        ref, offs, w_def, b_def, pred);
    // 3) h = relu(conv3x3(concat[ref, pred])) : 128 -> 64
    conv3x3_wmma_kernel<128, 64, 64, true, false>
        <<<cgrid, blk, ldsBytes(128), stream>>>(ref, pred, w_r1, b_r1,
                                                nullptr, hbuf);
    // 4) out = pred + conv3x3(h) : 64 -> 64
    conv3x3_wmma_kernel<64, 64, 64, false, true>
        <<<cgrid, blk, ldsBytes(64), stream>>>(hbuf, nullptr, w_r2, b_r2,
                                               pred, out);
}
